// GP_6141803233402
// MI455X (gfx1250) — matrix operand order repaired and checked by reading, not compiled, not run on hardware
//
#include <hip/hip_runtime.h>
#include <hip/hip_fp16.h>

// ---------------------------------------------------------------------------
// Problem constants (from reference): B=8, C=256, H=W=48 -> M=N=2304, D=64
// ---------------------------------------------------------------------------
#define GP_B   8
#define GP_C   256
#define GP_W   48
#define GP_M   2304      // 48*48
#define GP_D   64
#define GP_SIGMA 0.1f
#define GP_CG_ITERS 12
#define GP_CG_EPS 1e-9f

#ifndef __has_builtin
#define __has_builtin(x) 0
#endif
#if __has_builtin(__builtin_amdgcn_tensor_load_to_lds) && __has_builtin(__builtin_amdgcn_s_wait_tensorcnt)
#define GP_HAS_TDM 1
#else
#define GP_HAS_TDM 0
#endif

typedef __attribute__((ext_vector_type(16))) _Float16 v16h;
typedef __attribute__((ext_vector_type(8)))  _Float16 v8h;
typedef __attribute__((ext_vector_type(8)))  float    v8f;
typedef __attribute__((ext_vector_type(4)))  unsigned gp_v4u;
typedef __attribute__((ext_vector_type(4)))  int      gp_v4i;
typedef __attribute__((ext_vector_type(8)))  int      gp_v8i;

#define GP_WMMA(a, b, c) \
  __builtin_amdgcn_wmma_f32_16x16x32_f16(false, (a), false, (b), (short)0, (c), false, false)

static __device__ __forceinline__ v16h gp_cat(v8h a, v8h b) {
  return __builtin_shufflevector(a, b, 0,1,2,3,4,5,6,7,8,9,10,11,12,13,14,15);
}

// A-operand fragment: 16x32 f16 tile (rows row0.., K=k0..k0+31), row-major, ld
// elements. Contiguous K layout (matches B): lanes 0-15 hold K {0..15}, lanes
// 16-31 hold {16..31} -> two 16B loads per lane.
static __device__ __forceinline__ v16h gp_fragA(const _Float16* __restrict__ base,
                                                long row0, int k0, int ld, int lane) {
  const int m    = lane & 15;
  const int half = lane >> 4;
  const _Float16* p = base + (row0 + m) * (long)ld + (k0 + half * 16);
  v8h lo = *(const v8h*)(p);        // -> VGPR 0..3
  v8h hi = *(const v8h*)(p + 8);    // -> VGPR 4..7
  return gp_cat(lo, hi);
}

// B-operand fragment: 32x16 f16 tile whose column (col0+n) is stored as a row
// of a row-major matrix (computing X * Y^T). One contiguous 32B load per lane.
static __device__ __forceinline__ v16h gp_fragB(const _Float16* __restrict__ base,
                                                long col0, int k0, int ld, int lane) {
  const int n    = lane & 15;
  const int half = lane >> 4;
  return *(const v16h*)(base + (col0 + n) * (long)ld + (k0 + half * 16));
}

// ---------------------------------------------------------------------------
// Tensor Data Mover: DMA `rows` x `rowElems` f16 row-tile (row-major, stride =
// rowElems) from global into LDS at lds_dst. D# per CDNA5 ISA ch.8.
// ---------------------------------------------------------------------------
#if GP_HAS_TDM
static __device__ __forceinline__ void gp_tdm_load_rows(_Float16* lds_dst,
                                                        const _Float16* gsrc,
                                                        unsigned rows, unsigned rowElems) {
  const unsigned lds_addr = (unsigned)(unsigned long long)lds_dst;  // LDS byte offset
  const unsigned long long ga = (unsigned long long)gsrc;
  gp_v4u g0;
  g0[0] = 1u;                                          // count=1, user D#, no gather
  g0[1] = lds_addr;                                    // lds_addr [63:32]
  g0[2] = (unsigned)(ga & 0xffffffffu);                // global_addr [95:64]
  g0[3] = (unsigned)((ga >> 32) & 0x1ffffffu) | (2u << 30);  // addr[56:32] | type=2
  gp_v8i g1;
  g1[0] = (int)(1u << 16);                             // wg_mask=0, data_size=1 (2B)
  g1[1] = (int)((rowElems & 0xffffu) << 16);           // tensor_dim0 lo (bits 63:48)
  g1[2] = (int)(((rowElems >> 16) & 0xffffu) | ((rows & 0xffffu) << 16)); // dim0 hi | dim1 lo
  g1[3] = (int)(((rows >> 16) & 0xffffu) | ((rowElems & 0xffffu) << 16)); // dim1 hi | tile_dim0
  g1[4] = (int)(rows & 0xffffu);                       // tile_dim1 | tile_dim2=0
  g1[5] = (int)rowElems;                               // tensor_dim0_stride lo
  g1[6] = 0;                                           // stride0 hi | stride1 lo
  g1[7] = 0;                                           // stride1 hi
  gp_v4i gz = {0, 0, 0, 0};
#if __clang_major__ >= 23
  gp_v8i gz8 = {0, 0, 0, 0, 0, 0, 0, 0};
  __builtin_amdgcn_tensor_load_to_lds(g0, g1, gz, gz, gz8, 0);
#else
  __builtin_amdgcn_tensor_load_to_lds(g0, g1, gz, gz, 0);
#endif
}
#endif

// ---------------------------------------------------------------------------
// GEMM 1: Out[b][i][j] = exp( dot(Arows_i, Brows_j) - 1 ) + diag*(i==j), f16.
// Arows/Brows: (B, Mdim, Kdim=256) row-major f16 (L2-normalized rows).
// Block = 128 thr (4 waves as 2x2), block tile 64x64. The two 64x256 f16 row
// tiles (32KB each) are staged whole-K into LDS by the TDM; fragments then
// come from LDS (ds_load_b128).  grid = (Mdim/64, Mdim/64, B).
// ---------------------------------------------------------------------------
__global__ void gp_gemm_expk(const _Float16* __restrict__ Arows,
                             const _Float16* __restrict__ Brows,
                             _Float16* __restrict__ Out,
                             int Mdim, int Kdim, float diag) {
  __shared__ __align__(32) _Float16 shA[64 * GP_C];
  __shared__ __align__(32) _Float16 shB[64 * GP_C];
  const int lane = threadIdx.x & 31;
  const int wave = threadIdx.x >> 5;
  const int wi = wave >> 1, wj = wave & 1;
  const int b  = blockIdx.z;
  const long i0 = (long)blockIdx.y * 64;
  const long j0 = (long)blockIdx.x * 64;
  const _Float16* Ab = Arows + ((long)b * Mdim + i0) * Kdim;
  const _Float16* Bb = Brows + ((long)b * Mdim + j0) * Kdim;

#if GP_HAS_TDM
  if (wave == 0) {
    gp_tdm_load_rows(shA, Ab, 64, (unsigned)Kdim);
    gp_tdm_load_rows(shB, Bb, 64, (unsigned)Kdim);
    __builtin_amdgcn_s_wait_tensorcnt(0);
  }
#else
  for (int t = threadIdx.x * 8; t < 64 * Kdim; t += blockDim.x * 8) {
    *(v8h*)(shA + t) = *(const v8h*)(Ab + t);
    *(v8h*)(shB + t) = *(const v8h*)(Bb + t);
  }
#endif
  __syncthreads();

  const int ri = wi * 32, cj = wj * 32;
  v8f acc[2][2] = {};
  for (int k = 0; k < Kdim; k += 32) {
    v16h a0 = gp_fragA(shA, ri,      k, Kdim, lane);
    v16h a1 = gp_fragA(shA, ri + 16, k, Kdim, lane);
    v16h b0 = gp_fragB(shB, cj,      k, Kdim, lane);
    v16h b1 = gp_fragB(shB, cj + 16, k, Kdim, lane);
    acc[0][0] = GP_WMMA(a0, b0, acc[0][0]);
    acc[0][1] = GP_WMMA(a0, b1, acc[0][1]);
    acc[1][0] = GP_WMMA(a1, b0, acc[1][0]);
    acc[1][1] = GP_WMMA(a1, b1, acc[1][1]);
  }

  _Float16* Ob = Out + (long)b * Mdim * Mdim;
  const int n    = lane & 15;
  const int half = lane >> 4;
#pragma unroll
  for (int it = 0; it < 2; ++it)
#pragma unroll
    for (int jt = 0; jt < 2; ++jt) {
      const v8f A = acc[it][jt];
      const long r0 = i0 + ri + it * 16 + half * 8;
      const long cc = j0 + cj + jt * 16 + n;
#pragma unroll
      for (int r = 0; r < 8; ++r) {
        float v = __expf(A[r] - 1.0f);
        if ((r0 + r) == cc) v += diag;
        Ob[(r0 + r) * (long)Mdim + cc] = (_Float16)v;
      }
    }
}

// ---------------------------------------------------------------------------
// GEMM 2: Out[b][d][m] = sum_k Arows[b][d][k] * Brows[b][m][k], f32 out.
// Dm must be 64: each wave covers the FULL D extent (64x32 tile, 8 accums),
// so every Brows row (the 85MB A-matrix) is read exactly once per pass.
// Used for CG's AP^t = P^t A (A symmetric) and mu^t = Xsol^t Kxy^T.
// grid = (Nm/128, 1, B), block = 128 (4 waves, 32 cols each).
// ---------------------------------------------------------------------------
__global__ void gp_gemm_f16_f32out(const _Float16* __restrict__ Arows, // (B, 64, K)
                                   const _Float16* __restrict__ Brows, // (B, Nm, K)
                                   float* __restrict__ Out,            // (B, 64, Nm)
                                   int Dm, int Nm, int Kdim) {
  const int lane = threadIdx.x & 31;
  const int wave = threadIdx.x >> 5;
  const int b    = blockIdx.z;
  const long j0  = (long)blockIdx.x * 128 + wave * 32;
  const _Float16* Ab = Arows + (long)b * Dm * Kdim;
  const _Float16* Bb = Brows + (long)b * Nm * Kdim;

  v8f acc[4][2] = {};
  for (int k = 0; k < Kdim; k += 32) {
    if (k + 256 < Kdim) {  // gfx1250 global_prefetch_b8 of upcoming B rows
      __builtin_prefetch(Bb + (j0 + (lane & 15)) * (long)Kdim + k + 256, 0, 1);
    }
    v16h b0 = gp_fragB(Bb, j0,      k, Kdim, lane);
    v16h b1 = gp_fragB(Bb, j0 + 16, k, Kdim, lane);
    v16h a[4];
#pragma unroll
    for (int it = 0; it < 4; ++it) a[it] = gp_fragA(Ab, it * 16, k, Kdim, lane);
#pragma unroll
    for (int it = 0; it < 4; ++it) {
      acc[it][0] = GP_WMMA(a[it], b0, acc[it][0]);
      acc[it][1] = GP_WMMA(a[it], b1, acc[it][1]);
    }
  }

  float* Ob = Out + (long)b * Dm * Nm;
  const int n    = lane & 15;
  const int half = lane >> 4;
#pragma unroll
  for (int it = 0; it < 4; ++it)
#pragma unroll
    for (int jt = 0; jt < 2; ++jt) {
      const v8f A = acc[it][jt];
      const long r0 = it * 16 + half * 8;
      const long cc = j0 + jt * 16 + n;
#pragma unroll
      for (int r = 0; r < 8; ++r) Ob[(r0 + r) * (long)Nm + cc] = A[r];
    }
}

// ---------------------------------------------------------------------------
// Row-normalize features and transpose (B,C,HW) -> (B,HW,C) in f16.
// grid = (HW, B), block = 256 (= C).
// ---------------------------------------------------------------------------
__global__ void gp_normalize(const float* __restrict__ src,
                             _Float16* __restrict__ dst, int C, int HW) {
  const int b = blockIdx.y;
  const int m = blockIdx.x;
  const int c = threadIdx.x;
  float v = src[((long)b * C + c) * HW + m];
  __shared__ float red[256];
  red[c] = v * v;
  __syncthreads();
  for (int off = 128; off > 0; off >>= 1) {
    if (c < off) red[c] += red[c + off];
    __syncthreads();
  }
  const float rn = rsqrtf(red[0] + 1e-12f);
  dst[((long)b * HW + m) * C + c] = (_Float16)(v * rn);
}

// ---------------------------------------------------------------------------
// Positional encoding, stored transposed: F[b][d][m] = cos(8*pi*(w.g + b_d)).
// Base 16x16 grid bilinearly resized to 48x48 (align_corners=False, clamped).
// ---------------------------------------------------------------------------
static __device__ __forceinline__ float gp_grid1d(int i) {
  float c  = (i + 0.5f) * (16.0f / 48.0f) - 0.5f;
  float fl = floorf(c);
  float t  = c - fl;
  int   i0 = (int)fl;
  int   ia = min(max(i0, 0), 15);
  int   ib = min(max(i0 + 1, 0), 15);
  float va = (2.0f * ia + 1.0f) / 16.0f - 1.0f;
  float vb = (2.0f * ib + 1.0f) / 16.0f - 1.0f;
  return va + t * (vb - va);
}

__global__ void gp_fpos(const float* __restrict__ pos_w, const float* __restrict__ pos_b,
                        float* __restrict__ F, int D, int Mlen, int W, int Bn) {
  long i = (long)blockIdx.x * blockDim.x + threadIdx.x;
  if (i >= (long)D * Mlen) return;
  const int d = (int)(i / Mlen);
  const int m = (int)(i % Mlen);
  const int h = m / W, w = m % W;
  const float gx = gp_grid1d(w);
  const float gy = gp_grid1d(h);
  const float arg = pos_w[d * 2 + 0] * gx + pos_w[d * 2 + 1] * gy + pos_b[d];
  const float v = __cosf(25.132741228718345f * arg);   // 8*pi
  for (int b = 0; b < Bn; ++b) F[((long)b * D + d) * Mlen + m] = v;
}

// ---------------------------------------------------------------------------
// CG helpers. Vectors live transposed as (B, D, M); one block per (b,d) row.
// ---------------------------------------------------------------------------
__global__ void gp_cg_init(const float* __restrict__ F, float* __restrict__ X,
                           float* __restrict__ R, float* __restrict__ P,
                           _Float16* __restrict__ P16, long total) {
  long i = (long)blockIdx.x * blockDim.x + threadIdx.x;
  if (i >= total) return;
  const float f = F[i];
  X[i] = 0.0f; R[i] = f; P[i] = f; P16[i] = (_Float16)f;
}

__global__ void gp_rowdot(const float* __restrict__ a, const float* __restrict__ b,
                          float* __restrict__ out, int Mlen, float addeps) {
  const long row = blockIdx.x;
  __shared__ float red[256];
  float s = 0.0f;
  for (int m = threadIdx.x; m < Mlen; m += 256) {
    const long i = row * Mlen + m;
    s += a[i] * b[i];
  }
  red[threadIdx.x] = s;
  __syncthreads();
  for (int off = 128; off > 0; off >>= 1) {
    if (threadIdx.x < off) red[threadIdx.x] += red[threadIdx.x + off];
    __syncthreads();
  }
  if (threadIdx.x == 0) out[row] = red[0] + addeps;
}

__global__ void gp_cg_update1(float* __restrict__ X, float* __restrict__ R,
                              const float* __restrict__ P, const float* __restrict__ AP,
                              const float* __restrict__ rsold, const float* __restrict__ pAp,
                              float* __restrict__ rsnew, int Mlen) {
  const long row = blockIdx.x;
  const float alpha = rsold[row] / pAp[row];
  __shared__ float red[256];
  float s = 0.0f;
  for (int m = threadIdx.x; m < Mlen; m += 256) {
    const long i = row * Mlen + m;
    X[i] += alpha * P[i];
    const float r = R[i] - alpha * AP[i];
    R[i] = r;
    s += r * r;
  }
  red[threadIdx.x] = s;
  __syncthreads();
  for (int off = 128; off > 0; off >>= 1) {
    if (threadIdx.x < off) red[threadIdx.x] += red[threadIdx.x + off];
    __syncthreads();
  }
  if (threadIdx.x == 0) rsnew[row] = red[0];
}

__global__ void gp_cg_update2(float* __restrict__ P, _Float16* __restrict__ P16,
                              const float* __restrict__ R,
                              float* __restrict__ rsold, const float* __restrict__ rsnew,
                              int Mlen) {
  const long row = blockIdx.x;
  const float beta = rsnew[row] / (rsold[row] + GP_CG_EPS);
  for (int m = threadIdx.x; m < Mlen; m += 256) {
    const long i = row * Mlen + m;
    const float p = R[i] + beta * P[i];
    P[i] = p;
    P16[i] = (_Float16)p;
  }
  if (threadIdx.x == 0) rsold[row] = rsnew[row];
}

__global__ void gp_f32_to_f16(const float* __restrict__ in, _Float16* __restrict__ out, long total) {
  long i = (long)blockIdx.x * blockDim.x + threadIdx.x;
  if (i < total) out[i] = (_Float16)in[i];
}

// ---------------------------------------------------------------------------
// Host-side orchestration
// ---------------------------------------------------------------------------
extern "C" void kernel_launch(void* const* d_in, const int* in_sizes, int n_in,
                              void* d_out, int out_size, void* d_ws, size_t ws_size,
                              hipStream_t stream) {
  (void)in_sizes; (void)n_in; (void)out_size; (void)ws_size;
  const float* fA    = (const float*)d_in[0];   // (B, C, 48, 48)
  const float* fB    = (const float*)d_in[1];   // (B, C, 48, 48)
  const float* pos_w = (const float*)d_in[2];   // (64, 2)
  const float* pos_b = (const float*)d_in[3];   // (64,)

  const int B = GP_B, C = GP_C, M = GP_M, D = GP_D, W = GP_W;

  // Workspace carve-up (all 256B aligned)
  char* w = (char*)d_ws;
  auto carve = [&](size_t bytes) -> void* {
    void* p = (void*)w;
    w += (bytes + 255) & ~(size_t)255;
    return p;
  };
  _Float16* Yn    = (_Float16*)carve((size_t)B * M * C * 2);  // normalized fB rows
  _Float16* Xn    = (_Float16*)carve((size_t)B * M * C * 2);  // normalized fA rows
  _Float16* KA    = (_Float16*)carve((size_t)B * M * M * 2);  // A matrix, later Kxy
  float*    Fp    = (float*)carve((size_t)B * D * M * 4);     // Fpos^t
  float*    Xc    = (float*)carve((size_t)B * D * M * 4);
  float*    Rc    = (float*)carve((size_t)B * D * M * 4);
  float*    Pc    = (float*)carve((size_t)B * D * M * 4);
  float*    APc   = (float*)carve((size_t)B * D * M * 4);
  _Float16* P16   = (_Float16*)carve((size_t)B * D * M * 2);
  _Float16* Xs16  = (_Float16*)carve((size_t)B * D * M * 2);
  float*    rsold = (float*)carve((size_t)B * D * 4);
  float*    rsnew = (float*)carve((size_t)B * D * 4);
  float*    pap   = (float*)carve((size_t)B * D * 4);

  const long totDM = (long)B * D * M;

  // 1) Normalize features (and transpose to row-major rows, f16)
  gp_normalize<<<dim3(M, B), 256, 0, stream>>>(fB, Yn, C, M);
  gp_normalize<<<dim3(M, B), 256, 0, stream>>>(fA, Xn, C, M);

  // 2) Positional encoding F^t (B, D, M)
  {
    long tot = (long)D * M;
    gp_fpos<<<(unsigned)((tot + 255) / 256), 256, 0, stream>>>(pos_w, pos_b, Fp, D, M, W, B);
  }

  // 3) A = exp(Yn Yn^T - 1) + sigma*I, stored f16 (TDM-staged tiles)
  gp_gemm_expk<<<dim3(M / 64, M / 64, B), 128, 0, stream>>>(Yn, Yn, KA, M, C, GP_SIGMA);

  // 4) CG init: X=0, R=P=F, rsold = <R,R>
  gp_cg_init<<<(unsigned)((totDM + 255) / 256), 256, 0, stream>>>(Fp, Xc, Rc, Pc, P16, totDM);
  gp_rowdot<<<B * D, 256, 0, stream>>>(Rc, Rc, rsold, M, 0.0f);

  // 5) 12 CG iterations
  for (int it = 0; it < GP_CG_ITERS; ++it) {
    // AP^t = P^t * A  (A symmetric -> B-fragment from rows of A)
    gp_gemm_f16_f32out<<<dim3(M / 128, 1, B), 128, 0, stream>>>(P16, KA, APc, D, M, M);
    gp_rowdot<<<B * D, 256, 0, stream>>>(Pc, APc, pap, M, GP_CG_EPS);
    gp_cg_update1<<<B * D, 256, 0, stream>>>(Xc, Rc, Pc, APc, rsold, pap, rsnew, M);
    gp_cg_update2<<<B * D, 256, 0, stream>>>(Pc, P16, Rc, rsold, rsnew, M);
  }

  // 6) Xsol -> f16
  gp_f32_to_f16<<<(unsigned)((totDM + 255) / 256), 256, 0, stream>>>(Xc, Xs16, totDM);

  // 7) Kxy = exp(Xn Yn^T - 1) into the A buffer (A no longer needed)
  gp_gemm_expk<<<dim3(M / 64, M / 64, B), 128, 0, stream>>>(Xn, Yn, KA, M, C, 0.0f);

  // 8) mu^t = Xsol^t * Kxy^T  -> d_out laid out as (B, 64, 48, 48)
  gp_gemm_f16_f32out<<<dim3(M / 128, 1, B), 128, 0, stream>>>(Xs16, KA, (float*)d_out, D, M, M);
}
